// GPTTransformer_6356551598545
// MI455X (gfx1250) — compile-verified, matrix operand order checked by reading, not run on hardware
//
#include <hip/hip_runtime.h>
#include <hip/hip_bf16.h>

// GPT forward for MI455X (gfx1250): f16 WMMA GEMMs with async global->LDS
// B-tile staging (ASYNCcnt path). Weights pre-transposed to [N][K] so every
// WMMA fragment is per-lane contiguous (b128-width global/LDS accesses).

#define D_MODEL 512
#define N_HEAD  8
#define D_HEAD  64
#define FFN     2048
#define SEQ     1024
#define BATCH   4
#define NTOK    (BATCH * SEQ)          // 4096
#define VOCAB   50257
#define NLAYERS 6

typedef __attribute__((ext_vector_type(16))) _Float16 v16h;
typedef __attribute__((ext_vector_type(8)))  float    v8f;

// k-index inside a 16x32 f16 A/B fragment for element slot i of this lane.
// For i in 0..7 : k = hb + i      (hb = lane>=16 ? 8 : 0)   -> contiguous
// For i in 8..15: k = hb + 16 + (i-8)                       -> contiguous
__device__ __forceinline__ int kmap(int i, int lane) {
    return ((i & 8) ? 16 : 0) + ((lane & 16) ? 8 : 0) + (i & 7);
}

__device__ __forceinline__ void wait_asynccnt0() {
#if __has_builtin(__builtin_amdgcn_s_wait_asynccnt)
    __builtin_amdgcn_s_wait_asynccnt(0);
#else
    asm volatile("s_wait_asynccnt 0" ::: "memory");
#endif
}

// Async copy of 16 bytes global -> LDS (GLOBAL_LOAD_ASYNC_TO_LDS_B128, GV mode).
__device__ __forceinline__ void async_copy_b128(const void* gsrc, void* ldst) {
    unsigned  dso = (unsigned)(uintptr_t)ldst;       // flat->LDS: low 32 bits
    unsigned long long ga = (unsigned long long)(uintptr_t)gsrc;
    asm volatile("global_load_async_to_lds_b128 %0, %1, off"
                 :: "v"(dso), "v"(ga) : "memory");
}

// ---------------------------------------------------------------- convert
__global__ void f32_to_f16_kernel(const float* __restrict__ src,
                                  _Float16* __restrict__ dst, int n) {
    for (int i = blockIdx.x * blockDim.x + threadIdx.x; i < n;
         i += gridDim.x * blockDim.x)
        dst[i] = (_Float16)src[i];
}

// src is [K][N] f32 row-major; dst is [N][K] f16 (transposed).
__global__ void f32_to_f16_t_kernel(const float* __restrict__ src,
                                    _Float16* __restrict__ dst, int K, int N) {
    int total = K * N;
    for (int i = blockIdx.x * blockDim.x + threadIdx.x; i < total;
         i += gridDim.x * blockDim.x) {
        int k = i / N, n = i - k * N;
        dst[(size_t)n * K + k] = (_Float16)src[i];
    }
}

// ---------------------------------------------------------------- embed
__global__ void embed_kernel(const int* __restrict__ ids,
                             const float* __restrict__ emb,
                             const float* __restrict__ pe,
                             float* __restrict__ x) {
    int idx = blockIdx.x * blockDim.x + threadIdx.x;     // NTOK*D elements
    if (idx >= NTOK * D_MODEL) return;
    int token = idx >> 9;            // /512
    int d     = idx & (D_MODEL - 1);
    int s     = token & (SEQ - 1);
    int id    = ids[token];
    x[idx] = emb[(size_t)id * D_MODEL + d] * 22.627416998f   // sqrt(512)
           + pe[(size_t)s * D_MODEL + d];
}

// ---------------------------------------------------------------- layernorm
__global__ void ln_kernel(const float* __restrict__ x,
                          const float* __restrict__ g,
                          const float* __restrict__ b,
                          _Float16* __restrict__ out16) {
    int token = blockIdx.x * (blockDim.x >> 5) + (threadIdx.x >> 5);
    int lane  = threadIdx.x & 31;
    const float* xr = x + (size_t)token * D_MODEL;
    float v[16];
    float s = 0.f;
    #pragma unroll
    for (int i = 0; i < 16; ++i) { v[i] = xr[lane + i * 32]; s += v[i]; }
    #pragma unroll
    for (int off = 16; off; off >>= 1) s += __shfl_xor(s, off, 32);
    float mean = s * (1.0f / D_MODEL);
    float var = 0.f;
    #pragma unroll
    for (int i = 0; i < 16; ++i) { float d0 = v[i] - mean; var += d0 * d0; }
    #pragma unroll
    for (int off = 16; off; off >>= 1) var += __shfl_xor(var, off, 32);
    var *= (1.0f / D_MODEL);
    float inv = rsqrtf(var + 1e-5f);
    _Float16* orow = out16 + (size_t)token * D_MODEL;
    #pragma unroll
    for (int i = 0; i < 16; ++i) {
        int d0 = lane + i * 32;
        orow[d0] = (_Float16)((v[i] - mean) * inv * g[d0] + b[d0]);
    }
}

// ---------------------------------------------------------------- blocked GEMM
// C[M,N] = A[M,K](f16,row) * W^T where Bt is [N][K] (pre-transposed weights).
// Block = 128 thr (4 waves) -> 64x64 output tile. Per 32-wide k-step the block
// async-copies the 64n x 32k Bt tile into LDS (global_load_async_to_lds_b128,
// row-contiguous), then each wave reuses one A fragment across 4 WMMAs.
// All fragment reads are per-lane contiguous 16-byte chunks.
// Epilogues: epi 0: f16 out | epi 2: bias+erf-GELU f16 | epi 3: bias+residual f32
//            epi 4: f16 out transposed to vt[(b*8+h)*64+d][SEQ] (for attention V)
__global__ void gemm_f16_kernel(const _Float16* __restrict__ A,
                                const _Float16* __restrict__ Bt,
                                const float* __restrict__ bias,
                                float* __restrict__ resid,
                                _Float16* __restrict__ out16,
                                int M, int N, int K, int epi) {
    __shared__ _Float16 ldsB[64 * 32];            // 4 KB Bt tile: [n_local][k]
    int tid  = threadIdx.x;
    int lane = tid & 31;
    int wid  = tid >> 5;
    int rowbase  = blockIdx.x * 64 + wid * 16;
    int colblock = blockIdx.y * 64;
    int mlane = lane & 15;
    int hb = (lane & 16) ? 8 : 0;

    v8f acc[4] = {};
    const _Float16* Arow = A + (size_t)(rowbase + mlane) * K;

    for (int kb = 0; kb < K; kb += 32) {
        // cooperative async stage of Bt[colblock..+64][kb..kb+32]
        #pragma unroll
        for (int rep = 0; rep < 2; ++rep) {
            int ch = tid + rep * 128;             // 256 chunks of 16 B
            int nl = ch >> 2;                     // 0..63
            int cc = (ch & 3) * 8;                // half offset in row
            async_copy_b128(Bt + (size_t)(colblock + nl) * K + kb + cc,
                            ldsB + nl * 32 + cc);
        }
        wait_asynccnt0();
        __syncthreads();

        __builtin_prefetch(Arow + kb + 32, 0, 1);
        v16h a;
        #pragma unroll
        for (int i = 0; i < 16; ++i) a[i] = Arow[kb + kmap(i, lane)];

        #pragma unroll
        for (int nt = 0; nt < 4; ++nt) {
            const _Float16* brow = ldsB + (nt * 16 + mlane) * 32 + hb;
            v16h bfrag;
            #pragma unroll
            for (int i = 0; i < 8; ++i) {
                bfrag[i]     = brow[i];           // 16 B contiguous
                bfrag[i + 8] = brow[16 + i];      // 16 B contiguous
            }
            acc[nt] = __builtin_amdgcn_wmma_f32_16x16x32_f16(
                false, a, false, bfrag, (short)0, acc[nt], false, false);
        }
        __syncthreads();                          // LDS reuse next k-step
    }

    #pragma unroll
    for (int nt = 0; nt < 4; ++nt) {
        int n = colblock + nt * 16 + mlane;
        #pragma unroll
        for (int j = 0; j < 8; ++j) {
            int m = rowbase + hb + j;
            float v = acc[nt][j];
            if (bias) v += bias[n];
            if (epi == 2) v = 0.5f * v * (1.0f + erff(v * 0.7071067811865475f));
            if (epi == 3) {
                size_t o = (size_t)m * N + n;
                resid[o] = resid[o] + v;
            } else if (epi == 4) {
                // vt[(b*8+h)*64 + d][s]  with b=m>>10, s=m&1023, h=n>>6, d=n&63
                size_t o = (((size_t)(m >> 10) * 8 + (n >> 6)) * 64 + (n & 63))
                           * SEQ + (m & (SEQ - 1));
                out16[o] = (_Float16)v;
            } else {
                out16[(size_t)m * N + n] = (_Float16)v;
            }
        }
    }
}

// ---------------------------------------------------------------- attention
// One wave per (b, h, 16-query tile). Scores via WMMA -> 64KB LDS,
// full-row softmax (causal -10000 mask, matches reference), P*V via WMMA.
// V comes pre-transposed: vt[(b*8+h)*64+d][SEQ] -> contiguous B fragments.
__global__ void attn_kernel(const _Float16* __restrict__ q16,
                            const _Float16* __restrict__ k16,
                            const _Float16* __restrict__ vt16,
                            _Float16* __restrict__ ctx16) {
    __shared__ float s_s[16 * SEQ];               // 64 KB
    int lane = threadIdx.x & 31;
    int qt = blockIdx.x & 63;
    int bh = blockIdx.x >> 6;
    int b = bh >> 3, h = bh & 7;
    size_t rowoff = (size_t)b * SEQ * D_MODEL;
    int hd = h * D_HEAD;
    int mlane = lane & 15;
    int m0 = (lane & 16) ? 8 : 0;

    // Q fragments (16 x 64 split into two 16x32 slices); per-lane contiguous
    v16h aq0, aq1;
    {
        const _Float16* Qr = q16 + rowoff + (size_t)(qt * 16 + mlane) * D_MODEL + hd;
        #pragma unroll
        for (int i = 0; i < 16; ++i) {
            int kk = kmap(i, lane);
            aq0[i] = Qr[kk];
            aq1[i] = Qr[32 + kk];
        }
    }
    // Pass 1: scores = Q K^T / 8 with causal mask, into LDS
    for (int kt = 0; kt < 64; ++kt) {
        const _Float16* Kr = k16 + rowoff + (size_t)(kt * 16 + mlane) * D_MODEL + hd;
        v16h b0, b1;
        #pragma unroll
        for (int i = 0; i < 16; ++i) {
            int kk = kmap(i, lane);
            b0[i] = Kr[kk];
            b1[i] = Kr[32 + kk];
        }
        v8f c = {};
        c = __builtin_amdgcn_wmma_f32_16x16x32_f16(false, aq0, false, b0,
                                                   (short)0, c, false, false);
        c = __builtin_amdgcn_wmma_f32_16x16x32_f16(false, aq1, false, b1,
                                                   (short)0, c, false, false);
        int n_g = kt * 16 + mlane;
        #pragma unroll
        for (int j = 0; j < 8; ++j) {
            int m_g = qt * 16 + m0 + j;
            float s = c[j] * 0.125f;
            if (n_g > m_g) s = -10000.0f;
            s_s[(m0 + j) * SEQ + n_g] = s;
        }
    }
    __syncthreads();
    // Pass 2: softmax rows (lane pair handles one row: halves combined via shfl)
    float total;
    {
        int r = lane & 15;
        float* row = s_s + r * SEQ + (lane >> 4) * 512;
        float mx = -3.0e38f;
        for (int cc = 0; cc < 512; ++cc) mx = fmaxf(mx, row[cc]);
        mx = fmaxf(mx, __shfl_xor(mx, 16, 32));
        float sum = 0.f;
        for (int cc = 0; cc < 512; ++cc) {
            float pv = __expf(row[cc] - mx);
            row[cc] = pv;
            sum += pv;
        }
        total = sum + __shfl_xor(sum, 16, 32);    // every lane: sum of row (lane&15)
    }
    __syncthreads();
    // Pass 3: ctx = (P / l) * V   (P from LDS, V^T per-lane contiguous)
    for (int nt = 0; nt < 4; ++nt) {
        v8f c = {};
        const _Float16* Vd = vt16 + ((size_t)bh * 64 + nt * 16 + mlane) * SEQ;
        for (int kt2 = 0; kt2 < 32; ++kt2) {
            int kbase = kt2 * 32;
            v16h a, bb;
            #pragma unroll
            for (int i = 0; i < 16; ++i) {
                int kk = kmap(i, lane);
                a[i]  = (_Float16)s_s[mlane * SEQ + kbase + kk];
                bb[i] = Vd[kbase + kk];
            }
            c = __builtin_amdgcn_wmma_f32_16x16x32_f16(false, a, false, bb,
                                                       (short)0, c, false, false);
        }
        #pragma unroll
        for (int j = 0; j < 8; ++j) {
            int m = m0 + j;
            float inv_l = 1.0f / __shfl(total, m, 32);   // row-m sum lives in lane m
            ctx16[rowoff + (size_t)(qt * 16 + m) * D_MODEL + hd + nt * 16 + mlane] =
                (_Float16)(c[j] * inv_l);
        }
    }
}

// ---------------------------------------------------------------- logits GEMM
// out[M,V] = A[M,512] * emb^T ; emb16 is [V,512] so B access is per-lane
// contiguous in k. V edge guarded with zero-fill (EXEC stays all-ones).
__global__ void gemm_logits_kernel(const _Float16* __restrict__ A,
                                   const _Float16* __restrict__ E,
                                   float* __restrict__ out) {
    const int NT = (VOCAB + 15) / 16;             // 3142 column tiles
    int lane = threadIdx.x & 31;
    int wid  = threadIdx.x >> 5;
    int rowbase = blockIdx.x * 16;
    int coltile = blockIdx.y * 4 + wid;
    if (coltile >= NT) return;                    // whole-wave uniform exit
    int mlane = lane & 15;
    int n_g = coltile * 16 + mlane;
    bool nvalid = (n_g < VOCAB);

    v8f c = {};
    const _Float16* Arow = A + (size_t)(rowbase + mlane) * D_MODEL;
    const _Float16* Erow = E + (size_t)(nvalid ? n_g : 0) * D_MODEL;
    for (int kb = 0; kb < D_MODEL; kb += 32) {
        __builtin_prefetch(Erow + kb + 32, 0, 1);
        v16h a, bfrag;
        #pragma unroll
        for (int i = 0; i < 16; ++i) {
            int kk = kmap(i, lane);
            a[i]     = Arow[kb + kk];
            bfrag[i] = nvalid ? Erow[kb + kk] : (_Float16)0.0f;
        }
        c = __builtin_amdgcn_wmma_f32_16x16x32_f16(false, a, false, bfrag,
                                                   (short)0, c, false, false);
    }
    int m0 = (lane & 16) ? 8 : 0;
    if (!nvalid) return;
    #pragma unroll
    for (int j = 0; j < 8; ++j) {
        int m = rowbase + m0 + j;
        out[(size_t)m * VOCAB + n_g] = c[j];
    }
}

// ---------------------------------------------------------------- host
extern "C" void kernel_launch(void* const* d_in, const int* in_sizes, int n_in,
                              void* d_out, int out_size, void* d_ws, size_t ws_size,
                              hipStream_t stream) {
    const int*   ids   = (const int*)d_in[0];
    const float* emb   = (const float*)d_in[1];
    const float* pe    = (const float*)d_in[2];
    const float* wq    = (const float*)d_in[3];
    const float* wk    = (const float*)d_in[4];
    const float* wv    = (const float*)d_in[5];
    const float* wo    = (const float*)d_in[6];
    const float* wo_b  = (const float*)d_in[7];
    const float* w1    = (const float*)d_in[8];
    const float* b1    = (const float*)d_in[9];
    const float* w2    = (const float*)d_in[10];
    const float* b2    = (const float*)d_in[11];
    const float* ln1_g = (const float*)d_in[12];
    const float* ln1_b = (const float*)d_in[13];
    const float* ln2_g = (const float*)d_in[14];
    const float* ln2_b = (const float*)d_in[15];
    const float* lnf_g = (const float*)d_in[16];
    const float* lnf_b = (const float*)d_in[17];
    float* logits = (float*)d_out;

    char* p = (char*)d_ws;
    auto alloc = [&](size_t bytes) -> void* {
        void* r = (void*)p;
        p += (bytes + 255) & ~(size_t)255;
        return r;
    };
    float*    x      = (float*)   alloc((size_t)NTOK * D_MODEL * 4);
    _Float16* h16    = (_Float16*)alloc((size_t)NTOK * D_MODEL * 2);
    _Float16* q16    = (_Float16*)alloc((size_t)NTOK * D_MODEL * 2);
    _Float16* k16    = (_Float16*)alloc((size_t)NTOK * D_MODEL * 2);
    _Float16* vt16   = (_Float16*)alloc((size_t)NTOK * D_MODEL * 2);  // transposed V
    _Float16* ctx16  = (_Float16*)alloc((size_t)NTOK * D_MODEL * 2);
    _Float16* ff16   = (_Float16*)alloc((size_t)NTOK * FFN * 2);
    _Float16* wqT    = (_Float16*)alloc((size_t)D_MODEL * D_MODEL * 2);
    _Float16* wkT    = (_Float16*)alloc((size_t)D_MODEL * D_MODEL * 2);
    _Float16* wvT    = (_Float16*)alloc((size_t)D_MODEL * D_MODEL * 2);
    _Float16* woT    = (_Float16*)alloc((size_t)D_MODEL * D_MODEL * 2);
    _Float16* w1T    = (_Float16*)alloc((size_t)D_MODEL * FFN * 2);
    _Float16* w2T    = (_Float16*)alloc((size_t)FFN * D_MODEL * 2);
    _Float16* emb16  = (_Float16*)alloc((size_t)VOCAB * D_MODEL * 2);

    // --- weight conversion + transpose (deterministic, every call) ---
    auto cvtT = [&](const float* s, _Float16* d, int K, int N) {
        int n = K * N;
        int blocks = (n + 255) / 256; if (blocks > 4096) blocks = 4096;
        f32_to_f16_t_kernel<<<blocks, 256, 0, stream>>>(s, d, K, N);
    };
    cvtT(wq, wqT, D_MODEL, D_MODEL);
    cvtT(wk, wkT, D_MODEL, D_MODEL);
    cvtT(wv, wvT, D_MODEL, D_MODEL);
    cvtT(wo, woT, D_MODEL, D_MODEL);
    cvtT(w1, w1T, D_MODEL, FFN);
    cvtT(w2, w2T, FFN, D_MODEL);
    {
        int n = VOCAB * D_MODEL;
        f32_to_f16_kernel<<<4096, 256, 0, stream>>>(emb, emb16, n);
    }

    // --- embedding ---
    embed_kernel<<<(NTOK * D_MODEL + 255) / 256, 256, 0, stream>>>(ids, emb, pe, x);

    dim3 gB(128);
    dim3 gGemmD(NTOK / 64, D_MODEL / 64);         // 64 x 8
    dim3 gGemmF(NTOK / 64, FFN / 64);             // 64 x 32
    dim3 gLn(NTOK / 4);                           // 4 tokens per 128-thr block

    for (int layer = 0; layer < NLAYERS; ++layer) {
        ln_kernel<<<gLn, gB, 0, stream>>>(x, ln1_g, ln1_b, h16);
        gemm_f16_kernel<<<gGemmD, gB, 0, stream>>>(h16, wqT, nullptr, nullptr, q16,
                                                   NTOK, D_MODEL, D_MODEL, 0);
        gemm_f16_kernel<<<gGemmD, gB, 0, stream>>>(h16, wkT, nullptr, nullptr, k16,
                                                   NTOK, D_MODEL, D_MODEL, 0);
        gemm_f16_kernel<<<gGemmD, gB, 0, stream>>>(h16, wvT, nullptr, nullptr, vt16,
                                                   NTOK, D_MODEL, D_MODEL, 4);
        attn_kernel<<<BATCH * N_HEAD * (SEQ / 16), 32, 0, stream>>>(q16, k16, vt16, ctx16);
        gemm_f16_kernel<<<gGemmD, gB, 0, stream>>>(ctx16, woT, wo_b, x, nullptr,
                                                   NTOK, D_MODEL, D_MODEL, 3);
        ln_kernel<<<gLn, gB, 0, stream>>>(x, ln2_g, ln2_b, h16);
        gemm_f16_kernel<<<gGemmF, gB, 0, stream>>>(h16, w1T, b1, nullptr, ff16,
                                                   NTOK, FFN, D_MODEL, 2);
        gemm_f16_kernel<<<gGemmD, gB, 0, stream>>>(ff16, w2T, b2, x, nullptr,
                                                   NTOK, D_MODEL, FFN, 3);
    }
    ln_kernel<<<gLn, gB, 0, stream>>>(x, lnf_g, lnf_b, h16);

    const int NT = (VOCAB + 15) / 16;             // 3142
    dim3 gLog(NTOK / 16, (NT + 3) / 4);           // 256 x 786
    gemm_logits_kernel<<<gLog, gB, 0, stream>>>(h16, emb16, logits);
}
